// SpeGBM_59416577573152
// MI455X (gfx1250) — matrix operand order repaired and checked by reading, not compiled, not run on hardware
//
#include <hip/hip_runtime.h>

// ---------------- problem constants ----------------
namespace {
constexpr int kB = 4, kCH = 256, kH = 64, kW = 64;
constexpr int kTOK = 8, kGCN = 32, kPAD = 256, kDIN = 64, kDST = 16;
constexpr int kGROUPS = 4;
constexpr float kEPS = 1e-5f;
constexpr int kHW = kH * kW;        // 4096
constexpr int kM  = kB * kHW;       // 16384 spatial positions
constexpr int MB   = 8;             // spatial positions per workgroup
constexpr int ROWS = MB * kTOK;     // 64 (m,t) rows per block
constexpr int ROWS2 = 2 * ROWS;     // 128 (dir,m,t) rows
constexpr int NBLK = kM / MB;       // 2048 workgroups
constexpr int NPB  = kHW / MB;      // 512 workgroups per batch index
constexpr int NDBL = 48;            // padded x_dbl row stride (34 valid) -> branchless scatter
}

typedef __bf16 bf16_t;
typedef __attribute__((ext_vector_type(16))) __bf16 v16bf;
typedef __attribute__((ext_vector_type(8)))  float  v8f;

union VA { v16bf v; bf16_t e[16]; };
union VD { v8f v;   float  e[8];  };

static __device__ inline bf16_t f2bf(float f) { return (bf16_t)f; }   // RNE

// WMMA A operand index helper (contiguous k = 16*hh + i, same order as the B packing)
static __device__ inline int aKl(int i, int hh) {        // A 16x32: halfword i -> K
  return 16 * hh + i;
}

// =====================================================================
// Kernel 1: fully fused  xf->W_in -> bidirectional W_x -> dt/softplus ->
//           selective scan -> gelu gate -> W_out, plus per-block
//           GroupNorm partial sums. All intermediates live in LDS.
// Weights are pre-swizzled in LDS into B-operand register order so every
// WMMA B fetch is one contiguous 32-byte per-lane read (2x ds_load_b128).
// =====================================================================
__global__ __launch_bounds__(256)
void spegbm_fused(const float* __restrict__ x,
                  const float* __restrict__ W_in,
                  const float* __restrict__ W_x,
                  const float* __restrict__ W_dt,
                  const float* __restrict__ b_dt,
                  const float* __restrict__ A_logs,
                  const float* __restrict__ Ds,
                  const float* __restrict__ W_out,
                  float* __restrict__ y_pre,
                  float* __restrict__ partials)
{
  __shared__ __align__(32) bf16_t sXf[ROWS * kGCN];     // 4 KB   A operand, stage B
  __shared__ __align__(32) bf16_t pWi[1 * 8 * 32 * 16]; // 8 KB   W_in, B-operand order
  __shared__ __align__(32) bf16_t pWx[2 * 3 * 32 * 16]; // 6 KB   W_x,  B-operand order
  __shared__ __align__(32) bf16_t pWo[2 * 2 * 32 * 16]; // 4 KB   W_out,B-operand order
  __shared__ __align__(32) bf16_t sXp16[ROWS * kDIN];   // 8 KB   xp bf16 (A for C); reused as gelu-out
  __shared__ float  sXp[ROWS * kDIN];       // 16 KB  xp f32 (scan input u)
  __shared__ float  sZ [ROWS * kDIN];       // 16 KB  gate z
  __shared__ float  sDbl[ROWS2 * NDBL];     // 24 KB  x_dbl: dts|Bs|Cs (padded, branchless)
  __shared__ float  sYc[ROWS * kDIN];       // 16 KB  scan output; doubles as async staging
  __shared__ float  sYo[ROWS * kGCN];       // 8 KB   y @ W_out
  __shared__ float  sWdt[2 * kDIN];
  __shared__ float  sBdt[kDIN];
  __shared__ float  sNegA[kDIN * kDST];     // 4 KB   -exp(A_logs)
  __shared__ float  sDs[kDIN];
  __shared__ float  sRed[256 * 2];          // 2 KB   groupnorm partials

  const int tid  = threadIdx.x;
  const int lane = tid & 31;
  const int wv   = tid >> 5;
  const int ln16 = lane & 15;
  const int hh   = lane >> 4;

  const int blk = blockIdx.x;
  const int m0  = blk * MB;
  const int b   = m0 / kHW;
  const int hw0 = m0 % kHW;                 // MB divides kHW: block stays in one b

  // ---- async DMA: x tile (8 positions x 256 ch) HBM -> LDS, no VGPR data ----
  // IOFFSET is added to both LDS and global addresses (ISA 10.7), so one VDST
  // register covers both 16B halves. Staged raw f32 into sYc (dead until scan).
  {
    unsigned lds0 = (unsigned)(uintptr_t)&sYc[tid * MB];
    unsigned goff = (unsigned)(((b * kCH + tid) * kHW + hw0) * 4);
    asm volatile("global_load_async_to_lds_b128 %0, %1, %2\n\t"
                 "global_load_async_to_lds_b128 %0, %1, %2 offset:16"
                 :: "v"(lds0), "v"(goff), "s"(x) : "memory");
  }

  // ---- weights -> LDS, pre-swizzled into B-operand register order ----
  // pW[((kc*nCT+ct)*32 + lane)*16 + i] = W[(kc*32 + 16*(lane>>4) + i)*N + ct*16 + (lane&15)]
  for (int p = tid; p < 8 * 32 * 16; p += 256) {        // W_in: 1 kc x 8 ct, N=128
    int i = p & 15, ln = (p >> 4) & 31, ct = p >> 9;
    pWi[p] = f2bf(W_in[(16 * (ln >> 4) + i) * 128 + ct * 16 + (ln & 15)]);
  }
  for (int p = tid; p < 2 * 3 * 32 * 16; p += 256) {    // W_x: 2 kc x 3 ct, N=34
    int i = p & 15, ln = (p >> 4) & 31, t2 = p >> 9;
    int kc = t2 / 3, ct = t2 % 3;
    int K = kc * 32 + 16 * (ln >> 4) + i, col = ct * 16 + (ln & 15);
    pWx[p] = f2bf(col < 34 ? W_x[K * 34 + col] : 0.f);
  }
  for (int p = tid; p < 2 * 2 * 32 * 16; p += 256) {    // W_out: 2 kc x 2 ct, N=32
    int i = p & 15, ln = (p >> 4) & 31, t2 = p >> 9;
    int kc = t2 >> 1, ct = t2 & 1;
    pWo[p] = f2bf(W_out[(kc * 32 + 16 * (ln >> 4) + i) * kGCN + ct * 16 + (ln & 15)]);
  }
  for (int i = tid; i < 2 * kDIN; i += 256) sWdt[i] = W_dt[i];
  if (tid < kDIN) { sBdt[tid] = b_dt[tid]; sDs[tid] = Ds[tid]; }
  for (int i = tid; i < kDIN * kDST; i += 256) sNegA[i] = -expf(A_logs[i]);

  // ---- wait for this wave's async copies, convert own channel to bf16 A ----
  asm volatile("s_wait_asynccnt 0" ::: "memory");
  {
    int t = tid >> 5, g = tid & 31;        // c = tid = t*32+g
#pragma unroll
    for (int k = 0; k < MB; ++k)
      sXf[(k * kTOK + t) * kGCN + g] = f2bf(sYc[tid * MB + k]);
  }
  __syncthreads();

  // ---- Stage B: xz = xf @ W_in  (64x32)*(32x128), K=32 ----
  // wave wv: row-tile rt = wv&3, col-tiles ctBase..ctBase+3 (A loaded once,
  // reused by 4 WMMAs; xp/z split is a single wave-uniform region).
  {
    const int rt = wv & 3;
    const int ctBase = (wv >> 2) << 2;      // 0 (waves 0-3) or 4 (waves 4-7)
    const int row = rt * 16 + ln16;
    VA a;
#pragma unroll
    for (int i = 0; i < 16; ++i) a.e[i] = sXf[row * kGCN + aKl(i, hh)];
    VD d[4];
#pragma unroll
    for (int q = 0; q < 4; ++q) {
      VA bb;
      bb.v = *(const v16bf*)&pWi[((ctBase + q) * 32 + lane) * 16];
      v8f acc = {};
      acc = __builtin_amdgcn_wmma_f32_16x16x32_bf16(false, a.v, false, bb.v,
                                                    (short)0, acc, false, false);
      d[q].v = acc;
    }
    if (wv < 4) {                            // xp half: cols 0..63
#pragma unroll
      for (int q = 0; q < 4; ++q)
#pragma unroll
        for (int i = 0; i < 8; ++i) {
          int mrow = rt * 16 + i + 8 * hh;
          int col  = (ctBase + q) * 16 + ln16;
          sXp[mrow * kDIN + col]   = d[q].e[i];
          sXp16[mrow * kDIN + col] = f2bf(d[q].e[i]);
        }
    } else {                                 // z half: cols 64..127
#pragma unroll
      for (int q = 0; q < 4; ++q)
#pragma unroll
        for (int i = 0; i < 8; ++i) {
          int mrow = rt * 16 + i + 8 * hh;
          int col  = (ctBase + q) * 16 + ln16 - 64;
          sZ[mrow * kDIN + col] = d[q].e[i];
        }
    }
  }
  __syncthreads();

  // ---- Stage C: x_dbl = xs @ W_x  (128x64)*(64x34pad48) ----
  // wave wv owns row-tile wv; A (with reversed-token select) loaded once per
  // K-chunk, reused by 3 col-tile WMMAs.
  {
    const int rt  = wv;
    const int rr  = rt * 16 + ln16;
    const int dir = rr >> 6, r64 = rr & 63, lm = r64 >> 3, t = r64 & 7;
    const int srow = lm * kTOK + (dir ? (7 - t) : t);
    v8f acc[3] = {};
#pragma unroll
    for (int kc = 0; kc < 2; ++kc) {
      VA a;
#pragma unroll
      for (int i = 0; i < 16; ++i)
        a.e[i] = sXp16[srow * kDIN + kc * 32 + aKl(i, hh)];
#pragma unroll
      for (int ct = 0; ct < 3; ++ct) {
        VA bb;
        bb.v = *(const v16bf*)&pWx[((kc * 3 + ct) * 32 + lane) * 16];
        acc[ct] = __builtin_amdgcn_wmma_f32_16x16x32_bf16(false, a.v, false, bb.v,
                                                          (short)0, acc[ct], false, false);
      }
    }
#pragma unroll
    for (int ct = 0; ct < 3; ++ct) {
      VD d; d.v = acc[ct];
#pragma unroll
      for (int i = 0; i < 8; ++i)
        sDbl[(rt * 16 + i + 8 * hh) * NDBL + ct * 16 + ln16] = d.e[i];
    }
  }
  __syncthreads();

  // ---- init combined output with the skip term: 2*Ds*xp (both dirs) ----
  for (int i = tid; i < ROWS * kDIN; i += 256)
    sYc[i] = 2.f * sDs[i & 63] * sXp[i];
  __syncthreads();

  // ---- selective scan: 512 (m,d) pairs, fwd+bwd per pair, 16-state regs ----
  for (int p = tid; p < MB * kDIN; p += 256) {
    int lm = p >> 6, dch = p & 63;
    float An[kDST];
#pragma unroll
    for (int n = 0; n < kDST; ++n) An[n] = sNegA[dch * kDST + n];
    float w0 = sWdt[dch], w1 = sWdt[kDIN + dch];
    float bb2 = 2.f * sBdt[dch];                // reference adds b_dt twice
    for (int dir = 0; dir < 2; ++dir) {
      float hst[kDST];
#pragma unroll
      for (int n = 0; n < kDST; ++n) hst[n] = 0.f;
      int rbase = dir * ROWS + lm * kTOK;
      for (int t = 0; t < kTOK; ++t) {
        const float* rowd = &sDbl[(rbase + t) * NDBL];
        float val = rowd[0] * w0 + rowd[1] * w1 + bb2;
        float delta = (val > 20.f) ? val : log1pf(expf(val));   // softplus
        int ut = dir ? (7 - t) : t;
        float u  = sXp[(lm * kTOK + ut) * kDIN + dch];
        float du = delta * u;
        float y = 0.f;
#pragma unroll
        for (int n = 0; n < kDST; ++n) {
          float dA = __expf(delta * An[n]);    // native v_exp path, hot loop
          hst[n] = dA * hst[n] + du * rowd[2 + n];
          y += hst[n] * rowd[18 + n];
        }
        sYc[(lm * kTOK + ut) * kDIN + dch] += y;   // fwd at t, bwd at 7-t
      }
    }
  }
  __syncthreads();

  // ---- gelu gate (exact erf), pack bf16 A operand for stage F ----
  for (int i = tid; i < ROWS * kDIN; i += 256) {
    float z = sZ[i];
    float g = 0.5f * z * (1.f + erff(z * 0.70710678118654752f));
    sXp16[i] = f2bf(sYc[i] * g);                 // reuse sXp16 as sYg16
  }
  __syncthreads();

  // ---- Stage F: y @ W_out  (64x64)*(64x32), 2 K-chunks, 1 job/wave ----
  {
    const int rt = wv >> 1, ct = wv & 1;
    const int row = rt * 16 + ln16;
    v8f acc = {};
#pragma unroll
    for (int kc = 0; kc < 2; ++kc) {
      VA a, bb;
#pragma unroll
      for (int i = 0; i < 16; ++i)
        a.e[i] = sXp16[row * kDIN + kc * 32 + aKl(i, hh)];
      bb.v = *(const v16bf*)&pWo[((kc * 2 + ct) * 32 + lane) * 16];
      acc = __builtin_amdgcn_wmma_f32_16x16x32_bf16(false, a.v, false, bb.v,
                                                    (short)0, acc, false, false);
    }
    VD d; d.v = acc;
#pragma unroll
    for (int i = 0; i < 8; ++i)
      sYo[(rt * 16 + i + 8 * hh) * kGCN + ct * 16 + ln16] = d.e[i];
  }
  __syncthreads();

  // ---- store y_pre (NCHW) with 16B stores + per-block groupnorm partials ----
  {
    // thread tid owns channel c = tid; its MB=8 hw values are contiguous.
    float vals[MB];
    float s = 0.f, q = 0.f;
#pragma unroll
    for (int k = 0; k < MB; ++k) {
      float v = sYo[(k * kTOK + wv) * kGCN + lane];  // row (lm=k, t=wv), col=lane
      vals[k] = v; s += v; q += v * v;
    }
    float4* dst = (float4*)(y_pre + ((size_t)(b * kPAD + tid)) * kHW + hw0);
    dst[0] = make_float4(vals[0], vals[1], vals[2], vals[3]);
    dst[1] = make_float4(vals[4], vals[5], vals[6], vals[7]);
    sRed[tid * 2] = s; sRed[tid * 2 + 1] = q;
  }
  __syncthreads();
  if (tid < 8) {                 // group = c>>6 = tid>>6 -> 64 threads per group
    int grp = tid >> 1, w = tid & 1;
    float acc = 0.f;
    for (int l = 0; l < 64; ++l) acc += sRed[(grp * 64 + l) * 2 + w];
    partials[(size_t)blk * 8 + tid] = acc;
  }
}

// =====================================================================
// Kernel 2: deterministic tree reduction of per-block partials ->
//           per-(b,group) sum / sumsq.
// =====================================================================
__global__ __launch_bounds__(256)
void spegbm_reduce(const float* __restrict__ partials, float* __restrict__ accum)
{
  __shared__ float red[256];
  int o  = blockIdx.x;          // 0..31 = (b, group, sum|sumsq)
  int bb = o >> 3, k = o & 7;
  float s = 0.f;
  for (int j = threadIdx.x; j < NPB; j += 256)
    s += partials[((size_t)(bb * NPB + j)) * 8 + k];
  red[threadIdx.x] = s;
  __syncthreads();
  for (int st = 128; st > 0; st >>= 1) {
    if (threadIdx.x < st) red[threadIdx.x] += red[threadIdx.x + st];
    __syncthreads();
  }
  if (threadIdx.x == 0) accum[o] = red[0];
}

// =====================================================================
// Kernel 3: GroupNorm affine + residual, float4 vectorized.
// =====================================================================
__global__ __launch_bounds__(256)
void spegbm_finalize(const float* __restrict__ x,
                     const float* __restrict__ y_pre,
                     const float* __restrict__ accum,
                     const float* __restrict__ gn_w,
                     const float* __restrict__ gn_b,
                     float* __restrict__ out)
{
  size_t e4   = (size_t)blockIdx.x * blockDim.x + threadIdx.x;
  size_t base = e4 * 4;                        // 4 consecutive floats, same channel
  int bb  = (int)(base >> 20);                 // PAD*HW = 2^20
  int c   = (int)((base >> 12) & 255);         // HW = 2^12
  int grp = c >> 6;
  const float inv_n = 1.f / (float)((kPAD / kGROUPS) * kHW);   // 1/262144
  float s  = accum[bb * 8 + grp * 2 + 0];
  float q  = accum[bb * 8 + grp * 2 + 1];
  float mu = s * inv_n;
  float var = q * inv_n - mu * mu;
  float inv = rsqrtf(var + kEPS);
  float w   = gn_w[c] * inv;
  float bia = gn_b[c] - mu * w;
  float4 xv = *(const float4*)(x + base);
  float4 yv = *(const float4*)(y_pre + base);
  float4 o4;
  o4.x = xv.x + yv.x * w + bia;
  o4.y = xv.y + yv.y * w + bia;
  o4.z = xv.z + yv.z * w + bia;
  o4.w = xv.w + yv.w * w + bia;
  *(float4*)(out + base) = o4;
}

// =====================================================================
extern "C" void kernel_launch(void* const* d_in, const int* in_sizes, int n_in,
                              void* d_out, int out_size, void* d_ws, size_t ws_size,
                              hipStream_t stream)
{
  const float* x      = (const float*)d_in[0];
  const float* W_in   = (const float*)d_in[1];
  const float* W_x    = (const float*)d_in[2];
  const float* W_dt   = (const float*)d_in[3];
  const float* b_dt   = (const float*)d_in[4];
  const float* A_logs = (const float*)d_in[5];
  const float* Ds     = (const float*)d_in[6];
  const float* W_out  = (const float*)d_in[7];
  const float* gn_w   = (const float*)d_in[8];
  const float* gn_b   = (const float*)d_in[9];

  float* ws       = (float*)d_ws;
  float* partials = ws;                 // NBLK*8 = 16384 floats
  float* accum    = ws + NBLK * 8;      // 32 floats
  float* y_pre    = ws + 32768;         // 128KB-aligned region, 16 MB

  spegbm_fused   <<<NBLK, 256, 0, stream>>>(x, W_in, W_x, W_dt, b_dt, A_logs,
                                            Ds, W_out, y_pre, partials);
  spegbm_reduce  <<<32,   256, 0, stream>>>(partials, accum);
  spegbm_finalize<<<4096, 256, 0, stream>>>(x, y_pre, accum, gn_w, gn_b,
                                            (float*)d_out);
}
